// ShiftWiseConv_37014028157142
// MI455X (gfx1250) — matrix operand order repaired and checked by reading, not compiled, not run on hardware
//
#include <hip/hip_runtime.h>

// ---------------- types ----------------
typedef __bf16 v16bf __attribute__((ext_vector_type(16)));
typedef float  v8f   __attribute__((ext_vector_type(8)));

#define TAPS 169   // 13*13
#define C1   256
#define C2   256
#define HH   64
#define WW   64

__device__ __forceinline__ unsigned short f2bf(float f) {
  union { float f; unsigned u; } c; c.f = f;
  unsigned u = c.u;
  u += 0x7fffu + ((u >> 16) & 1u);   // round-to-nearest-even
  return (unsigned short)(u >> 16);
}

__device__ __forceinline__ unsigned pack_bf16x2(float a, float b) {
  union { __bf16 h[2]; unsigned u; } c;
  c.h[0] = (__bf16)a;                // fptrunc f32->bf16 (RNE)
  c.h[1] = (__bf16)b;
  return c.u;
}

// ---------------------------------------------------------------------------
// Pre-pass: repack W[co][ci][kh][kw] (f32) into bf16 WMMA A-fragments:
//   Wt[((tap*8 + cic)*16 + cob)*32 + lane][e]  (16 bf16 per lane = 32 B)
// A-fragment (16x32 bf16, MxK): lane L -> M = L%16;
//   lanes 0-15 hold K in {0..7, 16..23}; lanes 16-31 hold K in {8..15, 24..31}
// ---------------------------------------------------------------------------
__global__ __launch_bounds__(256) void repack_w_kernel(
    const float* __restrict__ Wsrc, unsigned short* __restrict__ Wt) {
  unsigned idx  = blockIdx.x * 256u + threadIdx.x;   // exact: TAPS*65536 threads
  unsigned e    = idx & 15u;
  unsigned lane = (idx >> 4) & 31u;
  unsigned cob  = (idx >> 9) & 15u;
  unsigned cic  = (idx >> 13) & 7u;
  unsigned tap  = idx >> 16;
  if (tap >= TAPS) return;
  unsigned g   = lane >> 4;
  unsigned m   = lane & 15u;
  unsigned k32 = g * 8u + (e & 7u) + ((e & 8u) << 1);  // within-lane K mapping
  unsigned ci  = cic * 32u + k32;
  unsigned co  = cob * 16u + m;
  float v = Wsrc[((size_t)co * C1 + ci) * TAPS + tap];
  Wt[idx] = f2bf(v);
}

// ---------------------------------------------------------------------------
// Main kernel: implicit-GEMM conv13x13 + BN(inference) + SiLU.
// Block = 8 waves -> 128(co) x 64(w) output tile of one (n, h) row.
// Double-buffered LDS B tile, staged fill (global loads -> WMMA -> LDS stores),
// one barrier per K-phase; B fragments preloaded 4-wide per ks group so the
// scheduler can clause the ds_loads and use progressive dscnt waits.
// grid = (2 co-superblocks, 512 = B*H rows).
// ---------------------------------------------------------------------------
__global__ __launch_bounds__(256) void conv_bn_silu_wmma(
    const float* __restrict__ x, const unsigned short* __restrict__ Wt,
    const float* __restrict__ gamma, const float* __restrict__ beta,
    const float* __restrict__ rmean, const float* __restrict__ rvar,
    float* __restrict__ out) {
  // B-fragment tile: [ks(4)][nb(4)][lane(32)][e(16)] bf16 = 4096 dwords = 16 KB
  __shared__ unsigned int lds_b[2][4096];
  __shared__ float lds_inv[128];
  __shared__ float lds_bias[128];

  const int t     = threadIdx.x;
  const int lane  = t & 31;
  const int wave  = t >> 5;
  const int bco   = blockIdx.x;      // 0..1 : 128-co superblock
  const int sp    = blockIdx.y;      // 0..511 : n*64 + h
  const int n_img = sp >> 6;
  const int h     = sp & 63;

  if (t < 128) {
    int co = (bco << 7) + t;
    float inv = gamma[co] * rsqrtf(rvar[co] + 1e-5f);
    lds_inv[t]  = inv;
    lds_bias[t] = beta[co] - rmean[co] * inv;
  }

  v8f acc[4] = {};   // 4 N-tiles of 16x16 f32

  // fill-phase mapping: thread -> output column fn (0..63), k-group fq (0..3)
  const int fn  = t & 63;
  const int fq  = t >> 6;
  const int n16 = fn & 15;
  const int nbb = fn >> 4;
  // dword index base inside a tile; per-kk remainder is a compile-time const
  const int tbase = (nbb << 8) + (n16 << 3) + fq;

  const float* xim = x + (size_t)n_img * (C1 * HH * WW);

  // valid kh range (rows falling inside the image): hs = h + kh - 6 in [0,64)
  const int kh0 = (h < 6) ? (6 - h) : 0;
  const int kh1 = (h > 57) ? (69 - h) : 12;
  const int np  = (kh1 - kh0 + 1) * 26;   // phases: (kh, r = kw*2 + kc)

  // base pointer for this thread's fill rows, given (kh, r)
  auto mk_ptr = [&](int kh, int r, bool& inb_o) -> const float* {
    const int kw = r >> 1;
    const int kc = r & 1;
    const int hs = h + kh - 6;
    const int wsrc = fn + kw - 6;
    const bool inb = (wsrc >= 0) && (wsrc < WW);
    inb_o = inb;
    return xim + (size_t)hs * WW + (size_t)kc * (128 * HH * WW)
               + (size_t)(fq * 2) * (HH * WW) + (inb ? wsrc : 0);
  };

  // ---- prologue: fill buffer 0 with phase 0 (kh0, r=0)
  {
    bool inb; const float* bp = mk_ptr(kh0, 0, inb);
    unsigned pk[16];
    #pragma unroll
    for (int kk = 0; kk < 16; ++kk) {
      float v0 = inb ? bp[(size_t)kk * 32768]        : 0.0f;
      float v1 = inb ? bp[(size_t)kk * 32768 + 4096] : 0.0f;
      pk[kk] = pack_bf16x2(v0, v1);
    }
    #pragma unroll
    for (int kk = 0; kk < 16; ++kk) {
      const int kconst = ((kk >> 2) << 10) | ((kk & 1) << 7) | (((kk >> 1) & 1) << 2);
      lds_b[0][tbase + kconst] = pk[kk];
    }
  }
  __syncthreads();

  int r_cur = 0, kh_cur = kh0;                 // phase p params (stepped, no div)
  for (int p = 0; p < np; ++p) {
    int r_nxt = r_cur + 1, kh_nxt = kh_cur;    // phase p+1
    if (r_nxt == 26) { r_nxt = 0; ++kh_nxt; }

    // ---- stage next phase's x rows into registers (hidden behind WMMAs)
    unsigned pk[16];
    const bool have_next = (p + 1 < np);
    if (have_next) {
      bool inb; const float* bp = mk_ptr(kh_nxt, r_nxt, inb);
      #pragma unroll
      for (int kk = 0; kk < 16; ++kk) {
        float v0 = inb ? bp[(size_t)kk * 32768]        : 0.0f;
        float v1 = inb ? bp[(size_t)kk * 32768 + 4096] : 0.0f;
        pk[kk] = pack_bf16x2(v0, v1);
      }
      if (p + 2 < np) {                        // distance-2 L2->WGP prefetch
        int r_p2 = r_nxt + 1, kh_p2 = kh_nxt;
        if (r_p2 == 26) { r_p2 = 0; ++kh_p2; }
        bool pi; const float* pf = mk_ptr(kh_p2, r_p2, pi);
        if (pi) __builtin_prefetch(pf, 0, 1);
      }
    }

    // ---- compute phase p: 16 WMMAs per wave; 4 independent B frags per ks
    {
      const int tap = kh_cur * 13 + (r_cur >> 1);
      const int kc  = r_cur & 1;
      const unsigned int* lb = lds_b[p & 1];
      const int cob = (bco << 3) + wave;       // 16-co block id (0..15)

      #pragma unroll
      for (int ks = 0; ks < 4; ++ks) {
        const size_t aoff =
            ((((size_t)tap * 8 + (kc << 2) + ks) * 16 + cob) * 32 + lane) * 16;
        const v16bf afrag = *reinterpret_cast<const v16bf*>(Wt + aoff);
        const unsigned int* lks = lb + ((ks << 10) + (lane << 3));
        const v16bf b0 = *reinterpret_cast<const v16bf*>(lks);
        const v16bf b1 = *reinterpret_cast<const v16bf*>(lks + (1 << 8));
        const v16bf b2 = *reinterpret_cast<const v16bf*>(lks + (2 << 8));
        const v16bf b3 = *reinterpret_cast<const v16bf*>(lks + (3 << 8));
        acc[0] = __builtin_amdgcn_wmma_f32_16x16x32_bf16(
            false, afrag, false, b0, (short)0, acc[0], false, false);
        acc[1] = __builtin_amdgcn_wmma_f32_16x16x32_bf16(
            false, afrag, false, b1, (short)0, acc[1], false, false);
        acc[2] = __builtin_amdgcn_wmma_f32_16x16x32_bf16(
            false, afrag, false, b2, (short)0, acc[2], false, false);
        acc[3] = __builtin_amdgcn_wmma_f32_16x16x32_bf16(
            false, afrag, false, b3, (short)0, acc[3], false, false);
      }
    }

    // ---- commit staged rows to the other buffer, then one barrier
    if (have_next) {
      unsigned int* lb = lds_b[(p + 1) & 1];
      #pragma unroll
      for (int kk = 0; kk < 16; ++kk) {
        const int kconst = ((kk >> 2) << 10) | ((kk & 1) << 7) | (((kk >> 1) & 1) << 2);
        lb[tbase + kconst] = pk[kk];
      }
    }
    __syncthreads();

    r_cur = r_nxt; kh_cur = kh_nxt;
  }

  // ---- epilogue: fused BN + SiLU, f32 NCHW stores
  // C/D layout: VGPR r -> M = r + 8*(lane/16); N = lane%16
  const int half = lane >> 4;
  const int cn16 = lane & 15;
  #pragma unroll
  for (int nb2 = 0; nb2 < 4; ++nb2) {
    const int w = (nb2 << 4) + cn16;
    #pragma unroll
    for (int r = 0; r < 8; ++r) {
      const int col = (wave << 4) + (half << 3) + r;   // co within 128-block
      const int co  = (bco << 7) + col;
      float y = acc[nb2][r] * lds_inv[col] + lds_bias[col];
      y = y * (1.0f / (1.0f + __expf(-y)));            // SiLU
      out[(((size_t)n_img * C2 + co) * HH + h) * WW + w] = y;
    }
  }
}

// ---------------------------------------------------------------------------
extern "C" void kernel_launch(void* const* d_in, const int* in_sizes, int n_in,
                              void* d_out, int out_size, void* d_ws, size_t ws_size,
                              hipStream_t stream) {
  const float* x     = (const float*)d_in[0];
  const float* W     = (const float*)d_in[1];
  const float* gamma = (const float*)d_in[2];
  const float* beta  = (const float*)d_in[3];
  const float* rmean = (const float*)d_in[4];
  const float* rvar  = (const float*)d_in[5];
  float* out = (float*)d_out;

  // d_ws: repacked bf16 weights, TAPS*8*16*512 elements = 22.1 MB
  unsigned short* Wt = (unsigned short*)d_ws;

  const int repack_threads = TAPS * 8 * 16 * 512;    // 11,075,584
  repack_w_kernel<<<repack_threads / 256, 256, 0, stream>>>(W, Wt);

  dim3 grid(2, 512);
  conv_bn_silu_wmma<<<grid, dim3(256), 0, stream>>>(x, Wt, gamma, beta,
                                                    rmean, rvar, out);
}
